// PCDAlignment_3_84275848282116
// MI455X (gfx1250) — matrix operand order repaired and checked by reading, not compiled, not run on hardware
//
#include <hip/hip_runtime.h>
#include <hip/hip_bf16.h>

// ---------------------------------------------------------------------------
// PCD alignment (EDVR) for MI455X / gfx1250.
// Dense convs + DCN channel reductions run on v_wmma_f32_16x16x32_f16
// (f16 operands, f32 accumulate); grouped convs (8 ch/group) run on VALU.
// Workspace use: ~128 MB (intermediates in f32, packed f16 weights at tail).
// ---------------------------------------------------------------------------

#define NFC   64
#define DGRP  8
#define NB    4
#define HH    96
#define WWW   96
#define HWSZ  (HH * WWW)
#define KTOT  576            // 64 cin * 9 kernel positions
#define KSTEPS 18            // 576 / 32
#define HALO  5              // max dilation -> zero halo for branch-free B fill
#define LDW   (WWW + 2 * HALO)   // 106 padded columns in LDS row stage

typedef _Float16 h8  __attribute__((ext_vector_type(8)));
typedef _Float16 h16 __attribute__((ext_vector_type(16)));
typedef float    f8  __attribute__((ext_vector_type(8)));

__device__ __forceinline__ float lrelu_f(float v) { return v >= 0.f ? v : 0.1f * v; }

// ---------------------------------------------------------------------------
// Channel-interleaved concat: out[:,2c]=a[:,c], out[:,2c+1]=b[:,c]
// ---------------------------------------------------------------------------
__global__ void cat_align_k(const float* __restrict__ a, const float* __restrict__ b,
                            float* __restrict__ out) {
  int idx = blockIdx.x * 256 + threadIdx.x;
  const int total = NB * NFC * HWSZ;
  if (idx >= total) return;
  int p  = idx % HWSZ;
  int r  = idx / HWSZ;
  int c  = r % NFC;
  int bb = r / NFC;
  out[((long)(bb * 2 * NFC + 2 * c    )) * HWSZ + p] = a[idx];
  out[((long)(bb * 2 * NFC + 2 * c + 1)) * HWSZ + p] = b[idx];
}

// ---------------------------------------------------------------------------
// Grouped 3x3 conv (8 groups, Cout=64, stride 1, pad=1, dil=1), optional lrelu.
// Only ~5% of total FLOPs -> plain VALU f32.
// ---------------------------------------------------------------------------
__global__ void gconv3x3_k(const float* __restrict__ x, const float* __restrict__ w,
                           const float* __restrict__ bias, float* __restrict__ out,
                           int cing, int do_lrelu) {
  int idx = blockIdx.x * 256 + threadIdx.x;
  const int total = NB * NFC * HWSZ;
  if (idx >= total) return;
  int px = idx % WWW;
  int r  = idx / WWW;
  int y  = r % HH; r /= HH;
  int oc = r % NFC;
  int bb = r / NFC;
  int g  = oc >> 3;
  int cin_total = cing * DGRP;
  float acc = bias[oc];
  for (int ci = 0; ci < cing; ++ci) {
    int ic = g * cing + ci;
    const float* xp = x + (long)(bb * cin_total + ic) * HWSZ;
    const float* wp = w + (long)(oc * cing + ci) * 9;
#pragma unroll
    for (int ky = 0; ky < 3; ++ky) {
      int yy = y + ky - 1;
      if (yy < 0 || yy >= HH) continue;
#pragma unroll
      for (int kx = 0; kx < 3; ++kx) {
        int xx = px + kx - 1;
        if (xx < 0 || xx >= WWW) continue;
        acc = fmaf(xp[yy * WWW + xx], wp[ky * 3 + kx], acc);
      }
    }
  }
  if (do_lrelu) acc = lrelu_f(acc);
  out[idx] = acc;
}

// ---------------------------------------------------------------------------
// Pack f32 OIHW (Cout,64,3,3) weights into the WMMA 16x32 f16 A-operand
// lane layout in this code's contiguous k order (same as the B fill): lane<16 -> halves 0..15 = K 0..15;
// lane>=16 -> K+16.  korder 0: K = kpos*64+cin (conv); 1: K = cin*9+kpos (dcn).
// ---------------------------------------------------------------------------
__global__ void pack_a_k(const float* __restrict__ w, _Float16* __restrict__ packed,
                         int cout, int mtiles, int korder) {
  int idx = blockIdx.x * 256 + threadIdx.x;
  int total = mtiles * KSTEPS * 32;
  if (idx >= total) return;
  int lane  = idx & 31;
  int kstep = (idx >> 5) % KSTEPS;
  int mtile = idx / (KSTEPS * 32);
  int m     = mtile * 16 + (lane & 15);
  int khalf = lane >> 4;
  _Float16* o = packed + (long)idx * 16;
  for (int j = 0; j < 16; ++j) {
    int kloc = khalf * 16 + j;
    int K = kstep * 32 + kloc;
    int cin, kpos;
    if (korder == 0) { kpos = K / NFC; cin = K % NFC; }
    else             { cin  = K / 9;   kpos = K % 9; }
    float v = (m < cout) ? w[((long)(m * NFC + cin)) * 9 + kpos] : 0.f;
    o[j] = (_Float16)v;
  }
}

// ---------------------------------------------------------------------------
// Dense 3x3 conv (Cin=64, Cout<=224) as implicit GEMM on WMMA.
// Block = 128 threads (4 waves), one (batch,row) per block.
// LDS stages 3 dilated input rows as [3][106][64] f16 (~40 KB) with a
// 5-column zero halo on each side, so the per-kstep B fill is two
// unconditional contiguous ds_load_b128 (no exec-mask branches in the loop).
// ---------------------------------------------------------------------------
__global__ void conv3x3_wmma_k(const float* __restrict__ x, const _Float16* __restrict__ packA,
                               const float* __restrict__ bias, float* __restrict__ out,
                               int cout, int mtiles, int dil) {
  __shared__ _Float16 ld[3 * LDW * NFC];
  int bid = blockIdx.x;
  int bb  = bid / HH;
  int y   = bid % HH;
  int tid = threadIdx.x;

  for (int e = tid; e < 3 * LDW * NFC; e += 128) {
    int cin  = e / (3 * LDW);
    int rem  = e % (3 * LDW);
    int rrow = rem / LDW;
    int xc   = rem % LDW;        // padded column
    int xw   = xc - HALO;        // real column
    int yy   = y + (rrow - 1) * dil;
    float v  = (yy >= 0 && yy < HH && xw >= 0 && xw < WWW)
                 ? x[((long)(bb * NFC + cin) * HH + yy) * WWW + xw] : 0.f;
    ld[(rrow * LDW + xc) * NFC + cin] = (_Float16)v;
  }
  __syncthreads();

  int wave  = tid >> 5;
  int lane  = tid & 31;
  int khalf = lane >> 4;
  int col   = lane & 15;
  const int ntiles = WWW / 16;
  int ttot = mtiles * ntiles;

  for (int t = wave; t < ttot; t += 4) {
    int mtile = t / ntiles;
    int nt    = t % ntiles;
    int px    = nt * 16 + col;
    const _Float16* ap = packA + (long)(mtile * KSTEPS) * 32 * 16 + (long)lane * 16;
    f8 acc = {};
    for (int kstep = 0; kstep < KSTEPS; ++kstep) {
      int kpos  = kstep >> 1;            // K = kpos*64 + cin, 32 K per step
      int ky    = kpos / 3, kx = kpos % 3;
      int cbase = (kstep & 1) * 32 + khalf * 16;
      int xc    = px + (kx - 1) * dil + HALO;      // always in [0, LDW)
      const h8* p = (const h8*)&ld[(ky * LDW + xc) * NFC + cbase];
      h8 lo = p[0], hi = p[1];
      h16 bm;
#pragma unroll
      for (int j = 0; j < 8; ++j) { bm[j] = lo[j]; bm[j + 8] = hi[j]; }
      h16 am = *(const h16*)(ap + (long)kstep * 32 * 16);
      acc = __builtin_amdgcn_wmma_f32_16x16x32_f16(false, am, false, bm, (short)0, acc,
                                                   false, false);
    }
#pragma unroll
    for (int i = 0; i < 8; ++i) {
      int m  = khalf * 8 + i;
      int oc = mtile * 16 + m;
      if (oc < cout)
        out[((long)(bb * cout + oc) * HH + y) * WWW + px] = acc[i] + bias[oc];
    }
  }
}

// ---------------------------------------------------------------------------
// Modulated deformable conv (DCNv2): bilinear sample * sigmoid(mask) into
// LDS [16 px][576 K] f16, then 64x576x16 GEMM on WMMA (4 M-tiles, 4 waves).
// om layout: ch 0..143 offsets ((g*9+k)*2 + dy/dx), ch 144..215 mask logits.
// ---------------------------------------------------------------------------
__global__ void dcn_wmma_k(const float* __restrict__ x, const float* __restrict__ om,
                           const _Float16* __restrict__ packW, const float* __restrict__ bias,
                           float* __restrict__ out, int dil, int do_lrelu) {
  __shared__ _Float16 smp[16 * KTOT];
  int bid = blockIdx.x;
  int nt  = bid % (WWW / 16);
  int r   = bid / (WWW / 16);
  int y   = r % HH;
  int bb  = r / HH;
  int tid = threadIdx.x;

  {  // sampling phase: thread = (pixel-in-tile, deformable group)
    int pl = tid >> 3;
    int g  = tid & 7;
    int xg = nt * 16 + pl;
    const float* omb = om + (long)bb * 216 * HWSZ;
#pragma unroll
    for (int k = 0; k < 9; ++k) {
      float dy = omb[(long)(g * 18 + 2 * k    ) * HWSZ + y * WWW + xg];
      float dx = omb[(long)(g * 18 + 2 * k + 1) * HWSZ + y * WWW + xg];
      float mk = omb[(long)(144 + g * 9 + k   ) * HWSZ + y * WWW + xg];
      mk = 1.f / (1.f + expf(-mk));
      float py  = (float)y  + (float)((k / 3) - 1) * (float)dil + dy;
      float pxx = (float)xg + (float)((k % 3) - 1) * (float)dil + dx;
      float y0f = floorf(py), x0f = floorf(pxx);
      float lyy = py - y0f, lxx = pxx - x0f;
      int iy0 = (int)y0f, ix0 = (int)x0f;
      float w00 = (1.f - lyy) * (1.f - lxx) * mk;
      float w01 = (1.f - lyy) * lxx * mk;
      float w10 = lyy * (1.f - lxx) * mk;
      float w11 = lyy * lxx * mk;
      bool by0 = (iy0 >= 0) && (iy0 < HH);
      bool by1 = (iy0 + 1 >= 0) && (iy0 + 1 < HH);
      bool bx0 = (ix0 >= 0) && (ix0 < WWW);
      bool bx1 = (ix0 + 1 >= 0) && (ix0 + 1 < WWW);
#pragma unroll
      for (int cc = 0; cc < 8; ++cc) {
        int c = g * 8 + cc;
        const float* xb = x + (long)(bb * NFC + c) * HWSZ;
        float v = 0.f;
        if (by0 && bx0) v = fmaf(w00, xb[iy0 * WWW + ix0],           v);
        if (by0 && bx1) v = fmaf(w01, xb[iy0 * WWW + ix0 + 1],       v);
        if (by1 && bx0) v = fmaf(w10, xb[(iy0 + 1) * WWW + ix0],     v);
        if (by1 && bx1) v = fmaf(w11, xb[(iy0 + 1) * WWW + ix0 + 1], v);
        smp[pl * KTOT + c * 9 + k] = (_Float16)v;  // K = c*9 + k
      }
    }
  }
  __syncthreads();

  int wave  = tid >> 5;
  int lane  = tid & 31;
  int khalf = lane >> 4;
  int col   = lane & 15;
  int mtile = wave;  // 4 waves <-> 4 output-channel tiles
  const _Float16* ap = packW + (long)(mtile * KSTEPS) * 32 * 16 + (long)lane * 16;
  f8 acc = {};
  for (int kstep = 0; kstep < KSTEPS; ++kstep) {
    h16 am = *(const h16*)(ap + (long)kstep * 32 * 16);
    const h8* p = (const h8*)&smp[col * KTOT + kstep * 32 + khalf * 16];
    h8 lo = p[0], hi = p[1];
    h16 bm;
#pragma unroll
    for (int j = 0; j < 8; ++j) { bm[j] = lo[j]; bm[j + 8] = hi[j]; }
    acc = __builtin_amdgcn_wmma_f32_16x16x32_f16(false, am, false, bm, (short)0, acc,
                                                 false, false);
  }
#pragma unroll
  for (int i = 0; i < 8; ++i) {
    int oc  = mtile * 16 + khalf * 8 + i;
    float v = acc[i] + bias[oc];
    if (do_lrelu) v = lrelu_f(v);
    out[((long)(bb * NFC + oc) * HH + y) * WWW + nt * 16 + col] = v;
  }
}

// ---------------------------------------------------------------------------
// Orchestration
// ---------------------------------------------------------------------------
extern "C" void kernel_launch(void* const* d_in, const int* in_sizes, int n_in,
                              void* d_out, int out_size, void* d_ws, size_t ws_size,
                              hipStream_t stream) {
  (void)in_sizes; (void)n_in; (void)out_size; (void)ws_size;
  const float* nbr = (const float*)d_in[0];
  const float* ref = (const float*)d_in[1];
#define FI(i) ((const float*)d_in[(i)])

  const long S64  = (long)NB * NFC * HWSZ;   // 2,359,296 floats
  const long S128 = 2 * S64;
  const long S216 = (long)NB * 216 * HWSZ;

  float* ws    = (float*)d_ws;
  float* cat0  = ws;                 // cat(nbr,ref), reused by all levels
  float* cat1  = cat0  + S128;
  float* om    = cat1  + S128;       // 216-ch offset/mask conv output
  float* tA    = om    + S216;
  float* tB    = tA    + S64;
  float* tC    = tB    + S64;
  float* tD    = tC    + S64;
  float* featP = tD    + S64;        // running aligned feature
  float* featR = featP + S64;        // raw dcn output before feat conv
  _Float16* hb = (_Float16*)(featR + S64);   // packed f16 weights
  const long PK_OM = 14L * KSTEPS * 32 * 16; // 129024 halves
  const long PK_D  = 4L  * KSTEPS * 32 * 16; //  36864 halves
  _Float16* pkOM0 = hb;
  _Float16* pkOM1 = hb + 1 * PK_OM;
  _Float16* pkOM2 = hb + 2 * PK_OM;
  _Float16* pkOM3 = hb + 3 * PK_OM;
  _Float16* pkD0  = hb + 4 * PK_OM;
  _Float16* pkD1  = pkD0 + PK_D;
  _Float16* pkD2  = pkD1 + PK_D;
  _Float16* pkD3  = pkD2 + PK_D;

  const int gE     = (int)((S64 + 255) / 256);   // elementwise / gconv grid
  const int CONV_G = NB * HH;                    // 384 blocks
  const int DCN_G  = NB * HH * (WWW / 16);       // 2304 blocks
  const int gbOM   = (14 * KSTEPS * 32 + 255) / 256;
  const int gbD    = (4  * KSTEPS * 32 + 255) / 256;

  // --- weight packing (f32 OIHW -> WMMA A-operand f16 layout) ---
  pack_a_k<<<gbOM, 256, 0, stream>>>(FI(6),  pkOM0, 216, 14, 0);
  pack_a_k<<<gbOM, 256, 0, stream>>>(FI(18), pkOM1, 216, 14, 0);
  pack_a_k<<<gbOM, 256, 0, stream>>>(FI(30), pkOM2, 216, 14, 0);
  pack_a_k<<<gbOM, 256, 0, stream>>>(FI(38), pkOM3, 216, 14, 0);
  pack_a_k<<<gbD,  256, 0, stream>>>(FI(8),  pkD0,  64,  4,  1);
  pack_a_k<<<gbD,  256, 0, stream>>>(FI(20), pkD1,  64,  4,  1);
  pack_a_k<<<gbD,  256, 0, stream>>>(FI(32), pkD2,  64,  4,  1);
  pack_a_k<<<gbD,  256, 0, stream>>>(FI(40), pkD3,  64,  4,  1);

  cat_align_k<<<gE, 256, 0, stream>>>(nbr, ref, cat0);

  // --- level 3 (dilation 5) ---
  gconv3x3_k<<<gE, 256, 0, stream>>>(cat0, FI(2), FI(3), tA, 16, 1);
  gconv3x3_k<<<gE, 256, 0, stream>>>(tA,   FI(4), FI(5), tB, 8,  1);   // tB = to_l3
  conv3x3_wmma_k<<<CONV_G, 128, 0, stream>>>(tB, pkOM0, FI(7), om, 216, 14, 5);
  dcn_wmma_k<<<DCN_G, 128, 0, stream>>>(nbr, om, pkD0, FI(9), featP, 5, 1);

  // --- level 2 (dilation 3) ---
  gconv3x3_k<<<gE, 256, 0, stream>>>(cat0, FI(10), FI(11), tA, 16, 1);
  cat_align_k<<<gE, 256, 0, stream>>>(tA, tB, cat1);
  gconv3x3_k<<<gE, 256, 0, stream>>>(cat1, FI(12), FI(13), tC, 16, 1);
  gconv3x3_k<<<gE, 256, 0, stream>>>(tC,   FI(14), FI(15), tD, 8,  1); // tD = to_l2
  conv3x3_wmma_k<<<CONV_G, 128, 0, stream>>>(tD, pkOM1, FI(19), om, 216, 14, 3);
  dcn_wmma_k<<<DCN_G, 128, 0, stream>>>(nbr, om, pkD1, FI(21), featR, 3, 0);
  cat_align_k<<<gE, 256, 0, stream>>>(featR, featP, cat1);
  gconv3x3_k<<<gE, 256, 0, stream>>>(cat1, FI(16), FI(17), featP, 16, 1);

  // --- level 1 (dilation 1; feat conv without lrelu) ---
  gconv3x3_k<<<gE, 256, 0, stream>>>(cat0, FI(22), FI(23), tA, 16, 1);
  cat_align_k<<<gE, 256, 0, stream>>>(tA, tD, cat1);
  gconv3x3_k<<<gE, 256, 0, stream>>>(cat1, FI(24), FI(25), tB, 16, 1);
  gconv3x3_k<<<gE, 256, 0, stream>>>(tB,   FI(26), FI(27), tC, 8,  1);
  conv3x3_wmma_k<<<CONV_G, 128, 0, stream>>>(tC, pkOM2, FI(31), om, 216, 14, 1);
  dcn_wmma_k<<<DCN_G, 128, 0, stream>>>(nbr, om, pkD2, FI(33), featR, 1, 0);
  cat_align_k<<<gE, 256, 0, stream>>>(featR, featP, cat1);
  gconv3x3_k<<<gE, 256, 0, stream>>>(cat1, FI(28), FI(29), featP, 16, 0);

  // --- cascade (dilation 1, lrelu on output) ---
  cat_align_k<<<gE, 256, 0, stream>>>(featP, ref, cat1);
  gconv3x3_k<<<gE, 256, 0, stream>>>(cat1, FI(34), FI(35), tA, 16, 1);
  gconv3x3_k<<<gE, 256, 0, stream>>>(tA,   FI(36), FI(37), tB, 8,  1);
  conv3x3_wmma_k<<<CONV_G, 128, 0, stream>>>(tB, pkOM3, FI(39), om, 216, 14, 1);
  dcn_wmma_k<<<DCN_G, 128, 0, stream>>>(featP, om, pkD3, FI(41), (float*)d_out, 1, 1);
#undef FI
}
